// SimpleBatchAttention_18202071400990
// MI455X (gfx1250) — matrix operand order repaired and checked by reading, not compiled, not run on hardware
//
#include <hip/hip_runtime.h>

// ---------------------------------------------------------------------------
// Problem constants (match reference):  B=16 S=4 N=2048 D=128 H=16 M=4096
// ---------------------------------------------------------------------------
#define Bx 16
#define Sx 4
#define Nx 2048
#define Dx 128
#define Hx 16
#define Mx_CACHE 4096
#define BH (Bx * Hx)        // 256
#define ROWS (Bx * Sx)      // 64 query rows

typedef __attribute__((ext_vector_type(16))) __bf16 v16bf;
typedef __attribute__((ext_vector_type(8)))  float  v8f;
typedef int i32x4 __attribute__((vector_size(16)));

#define DEVI __device__ __forceinline__
#define AS1 __attribute__((address_space(1)))
#define AS3 __attribute__((address_space(3)))

DEVI __bf16 bfbits(unsigned short u) {
    union { unsigned short u; __bf16 h; } c; c.u = u; return c.h;
}
DEVI __bf16 f2bf(float f) { return (__bf16)f; }          // native v_cvt_*bf16_f32
DEVI unsigned short f2bfu(float f) {
    union { unsigned short u; __bf16 h; } c; c.h = (__bf16)f; return c.u;
}

// Float atomic-max via signed/unsigned integer ordering trick (init to -inf).
DEVI void atomicMaxF(float* addr, float val) {
    if (val >= 0.0f) atomicMax((int*)addr, __float_as_int(val));
    else             atomicMin((unsigned int*)addr, __float_as_uint(val));
}

DEVI v8f wmma_bf16(v16bf a, v16bf b, v8f c) {
    return __builtin_amdgcn_wmma_f32_16x16x32_bf16(
        /*neg_a=*/false, a, /*neg_b=*/false, b,
        /*c_mod=*/(short)0, c, /*reuse_a=*/false, /*reuse_b=*/false);
}

// --------------------- async-to-LDS support (guarded) ----------------------
#if __has_builtin(__builtin_amdgcn_global_load_async_to_lds_b128)
#define HAS_ASYNC_LDS 1
DEVI void async_ld_b128(const float* g, float* l) {
    __builtin_amdgcn_global_load_async_to_lds_b128(
        (AS1 i32x4*)g, (AS3 i32x4*)l, 0, 0);
}
#else
#define HAS_ASYNC_LDS 0
#endif

#if __has_builtin(__builtin_amdgcn_s_wait_asynccnt)
#define WAIT_ASYNC(n) __builtin_amdgcn_s_wait_asynccnt(n)
#else
#define WAIT_ASYNC(n) asm volatile("s_wait_asynccnt %0" :: "n"(n) : "memory")
#endif

// ---------------------------------------------------------------------------
// Kernel 0: RMSNorm (no eps, no scale) -> bf16 bits in workspace
// ---------------------------------------------------------------------------
__global__ void rmsnorm_kernel(const float* __restrict__ X, unsigned short* __restrict__ Xn) {
    int row = blockIdx.x, tid = threadIdx.x;
    const float* x = X + (size_t)row * Nx;
    float ss = 0.0f;
    for (int i = tid; i < Nx; i += 256) { float v = x[i]; ss += v * v; }
    __shared__ float red[256];
    red[tid] = ss; __syncthreads();
    for (int s = 128; s > 0; s >>= 1) { if (tid < s) red[tid] += red[tid + s]; __syncthreads(); }
    float r = rsqrtf(red[0] * (1.0f / (float)Nx));
    for (int i = tid; i < Nx; i += 256) Xn[(size_t)row * Nx + i] = f2bfu(x[i] * r);
}

__global__ void init_mx_kernel(float* __restrict__ Mxv) {
    int i = blockIdx.x * blockDim.x + threadIdx.x;
    if (i < BH * Sx) Mxv[i] = -3.4e38f;
}

// ---------------------------------------------------------------------------
// Kernel 1: QKV GEMM  Y[64x2048] = Xn @ W  (bf16 WMMA, f32 accumulate)
// grid (128 n-tiles, 3 proj), block 128 (4 waves; wave w -> rows 16w..16w+15)
// ---------------------------------------------------------------------------
__global__ void qkv_kernel(const unsigned short* __restrict__ Xn,
                           const float* __restrict__ Wq, const float* __restrict__ Wk,
                           const float* __restrict__ Wv,
                           float* __restrict__ Qws, float* __restrict__ Kn, float* __restrict__ Vn) {
    int proj = blockIdx.y;
    int n0   = blockIdx.x * 16;
    const float* W  = (proj == 0) ? Wq : (proj == 1 ? Wk : Wv);
    float* dst      = (proj == 0) ? Qws : (proj == 1 ? Kn : Vn);

    int tid = threadIdx.x, wave = tid >> 5, lane = tid & 31;
    int m0 = wave * 16;
    int mrow = m0 + (lane & 15);
    int koff = (lane >> 4) << 3;

    v8f c = {0.f,0.f,0.f,0.f,0.f,0.f,0.f,0.f};
    for (int k0 = 0; k0 < Nx; k0 += 32) {
        v16bf a, bw;
        const unsigned short* arow = Xn + (size_t)mrow * Nx + k0 + koff;
        #pragma unroll
        for (int i = 0; i < 16; ++i) a[i] = bfbits(arow[((i >> 3) << 4) + (i & 7)]);
        const float* wr = W + (size_t)(k0 + koff) * Nx + n0 + (lane & 15);   // column n = lane%16, k in table order
        #pragma unroll
        for (int i = 0; i < 16; ++i) bw[i] = f2bf(wr[(size_t)(((i >> 3) << 4) + (i & 7)) * Nx]);
        c = wmma_bf16(a, bw, c);
    }
    int roff = (lane >> 4) << 3;
    int col = n0 + (lane & 15);
    int h = col >> 7, d = col & 127;
    #pragma unroll
    for (int v = 0; v < 8; ++v) {
        int row = m0 + v + roff;               // row = b*4 + s
        int b = row >> 2, s = row & 3;
        dst[(((size_t)b * Hx + h) * Sx + s) * Dx + d] = c[v];   // [bh][s][d]
    }
}

// ---------------------------------------------------------------------------
// Kernel 2: scores  S[bh][q][m] = Q . K  (rows >= P redirected to fresh K)
// grid (4 segs, 256 bh), block 128 (4 waves; wave covers 256 cache rows)
// ---------------------------------------------------------------------------
__global__ void scores_kernel(const float* __restrict__ Qws, const float* __restrict__ cacheK,
                              const float* __restrict__ Kn, const int* __restrict__ Pp,
                              float* __restrict__ Sws, float* __restrict__ Mxv) {
    int bh = blockIdx.y, seg = blockIdx.x;
    int tid = threadIdx.x, wave = tid >> 5, lane = tid & 31;
    int q = lane & 15, koff = (lane >> 4) << 3;
    int P = *Pp;

    // Preload Q fragments (rows >= S are zero padding)
    v16bf aq[4];
    #pragma unroll
    for (int f = 0; f < 4; ++f) {
        #pragma unroll
        for (int i = 0; i < 16; ++i) {
            int k = f * 32 + ((i >> 3) << 4) + koff + (i & 7);
            float qv = (q < Sx) ? Qws[((size_t)bh * Sx + q) * Dx + k] : 0.0f;
            aq[f][i] = f2bf(qv);
        }
    }

    float lmax[4] = {-3.4e38f, -3.4e38f, -3.4e38f, -3.4e38f};
    int base = seg * 1024 + wave * 256;

    for (int t = 0; t < 16; ++t) {
        int m0 = base + t * 16;
        if (m0 + 16 + 15 < Mx_CACHE)
            __builtin_prefetch(cacheK + ((size_t)bh * Mx_CACHE + m0 + 16 + (lane & 15)) * Dx, 0, 0);
        v8f c = {0.f,0.f,0.f,0.f,0.f,0.f,0.f,0.f};
        #pragma unroll
        for (int f = 0; f < 4; ++f) {
            v16bf bk;
            int r = m0 + (lane & 15);                      // B: N = lane%16
            const float* src = (r >= P) ? (Kn + ((size_t)bh * Sx + (r - P)) * Dx)
                                        : (cacheK + ((size_t)bh * Mx_CACHE + r) * Dx);
            #pragma unroll
            for (int i = 0; i < 16; ++i) {
                int d = f * 32 + ((i >> 3) << 4) + koff + (i & 7);   // B: K in table order
                bk[i] = f2bf(src[d]);
            }
            c = wmma_bf16(aq[f], bk, c);
        }
        if (lane < 16) {                                   // rows 0..3 live in lanes 0..15
            #pragma unroll
            for (int v = 0; v < 4; ++v) {
                float sv = c[v];
                lmax[v] = fmaxf(lmax[v], sv);
                Sws[((size_t)bh * Sx + v) * Mx_CACHE + m0 + lane] = sv;
            }
        }
    }
    #pragma unroll
    for (int v = 0; v < 4; ++v) {
        float m = lmax[v];
        m = fmaxf(m, __shfl_xor(m, 1, 32));
        m = fmaxf(m, __shfl_xor(m, 2, 32));
        m = fmaxf(m, __shfl_xor(m, 4, 32));
        m = fmaxf(m, __shfl_xor(m, 8, 32));
        if (lane == 0) atomicMaxF(&Mxv[bh * Sx + v], m);
    }
}

// ---------------------------------------------------------------------------
// Kernel 3: softmax denominators  L[bh][q] = sum_m exp(S - max)
// ---------------------------------------------------------------------------
__global__ void sums_kernel(const float* __restrict__ Sws, const float* __restrict__ Mxv,
                            float* __restrict__ L) {
    int bh = blockIdx.x, tid = threadIdx.x;
    __shared__ float acc[4];
    if (tid < 4) acc[tid] = 0.0f;
    __syncthreads();
    float mx[4];
    #pragma unroll
    for (int qv = 0; qv < 4; ++qv) mx[qv] = Mxv[bh * Sx + qv];
    float s[4] = {0.f, 0.f, 0.f, 0.f};
    for (int m = tid; m < Mx_CACHE; m += 256) {
        #pragma unroll
        for (int qv = 0; qv < 4; ++qv)
            s[qv] += __expf(Sws[((size_t)bh * Sx + qv) * Mx_CACHE + m] - mx[qv]);
    }
    #pragma unroll
    for (int qv = 0; qv < 4; ++qv) atomicAdd(&acc[qv], s[qv]);
    __syncthreads();
    if (tid < 4) L[bh * Sx + tid] = acc[tid];
}

// ---------------------------------------------------------------------------
// Kernel 4: O = softmax(S) @ V   (rows >= P redirected to fresh V)
// grid 256 (bh), block 128 (4 waves; wave covers 1024 cache rows).
// V is staged through LDS with double-buffered GLOBAL_LOAD_ASYNC_TO_LDS_B128:
// each wave copies its 32x128f chunk (32 issues x 512B contiguous) while WMMA
// consumes the previous chunk; ASYNCcnt in-order completion => wait(32) means
// the older chunk has fully landed. Rows >= P are patched from fresh V.
// ---------------------------------------------------------------------------
#define VROW 132   // padded LDS row stride in floats (16B aligned, banks spread)

__global__ void pv_kernel(const float* __restrict__ Sws, const float* __restrict__ Mxv,
                          const float* __restrict__ L, const float* __restrict__ cacheV,
                          const float* __restrict__ Vn, const int* __restrict__ Pp,
                          float* __restrict__ out) {
    int bh = blockIdx.x;
    int tid = threadIdx.x, wave = tid >> 5, lane = tid & 31;
    int q = lane & 15, koff = (lane >> 4) << 3;
    int P = *Pp;

    __shared__ float o_sh[Sx * Dx];                        // 4 x 128
    for (int i = tid; i < Sx * Dx; i += 128) o_sh[i] = 0.0f;
    __syncthreads();

    float mx[4], il[4];
    #pragma unroll
    for (int qv = 0; qv < 4; ++qv) {
        mx[qv] = Mxv[bh * Sx + qv];
        il[qv] = 1.0f / L[bh * Sx + qv];
    }

    v8f zero = {0.f,0.f,0.f,0.f,0.f,0.f,0.f,0.f};
    v8f acc[8];
    #pragma unroll
    for (int t = 0; t < 8; ++t) acc[t] = zero;

#if HAS_ASYNC_LDS
    __shared__ float stage[4][2][32 * VROW];               // per-wave double buffer
    const float* vbase = cacheV + (size_t)bh * Mx_CACHE * Dx;

    // Issue one 32-row chunk: instruction u copies cache row (cb+u); lane l
    // covers its 16B column segment -> each issue is one contiguous 512B line.
    auto issue_chunk = [&](int ch, float* st) {
        int cb = wave * 1024 + ch * 32;
        const float* g = vbase + (size_t)cb * Dx + lane * 4;
        float* l = st + lane * 4;
        #pragma unroll
        for (int u = 0; u < 32; ++u)
            async_ld_b128(g + u * Dx, l + u * VROW);
    };

    issue_chunk(0, &stage[wave][0][0]);

    for (int ch = 0; ch < 32; ++ch) {
        float* cur = &stage[wave][ch & 1][0];
        float* nxt = &stage[wave][(ch & 1) ^ 1][0];

        // A fragment: normalized probabilities (exp on the trans pipe)
        v16bf a;
        #pragma unroll
        for (int i = 0; i < 16; ++i) {
            int k = ((i >> 3) << 4) + koff + (i & 7);
            float p = 0.0f;
            if (q < Sx)
                p = __expf(Sws[((size_t)bh * Sx + q) * Mx_CACHE + wave * 1024 + ch * 32 + k]
                           - mx[q]) * il[q];
            a[i] = f2bf(p);
        }

        if (ch + 1 < 32) { issue_chunk(ch + 1, nxt); WAIT_ASYNC(32); }
        else             { WAIT_ASYNC(0); }

        // Patch rows belonging to the freshly written KV slots (r >= P).
        int r = wave * 1024 + ch * 32 + lane;
        if (r >= P) {
            const float* src = Vn + ((size_t)bh * Sx + (r - P)) * Dx;
            #pragma unroll
            for (int u = 0; u < Dx; u += 4) {
                cur[lane * VROW + u + 0] = src[u + 0];
                cur[lane * VROW + u + 1] = src[u + 1];
                cur[lane * VROW + u + 2] = src[u + 2];
                cur[lane * VROW + u + 3] = src[u + 3];
            }
        }

        #pragma unroll
        for (int t = 0; t < 8; ++t) {
            v16bf bv;                                       // B: K=table-order row, N=t*16+lane%16
            #pragma unroll
            for (int i = 0; i < 16; ++i) bv[i] = f2bf(cur[(((i >> 3) << 4) + koff + (i & 7)) * VROW + t * 16 + (lane & 15)]);
            acc[t] = wmma_bf16(a, bv, acc[t]);
        }
    }
#else
    for (int ch = 0; ch < 32; ++ch) {
        int cb = wave * 1024 + ch * 32;
        v16bf a;
        #pragma unroll
        for (int i = 0; i < 16; ++i) {
            int k = ((i >> 3) << 4) + koff + (i & 7);
            float p = 0.0f;
            if (q < Sx)
                p = __expf(Sws[((size_t)bh * Sx + q) * Mx_CACHE + cb + k] - mx[q]) * il[q];
            a[i] = f2bf(p);
        }
        int r = cb + lane;
        const float* vrow = (r >= P) ? (Vn + ((size_t)bh * Sx + (r - P)) * Dx)
                                     : (cacheV + ((size_t)bh * Mx_CACHE + r) * Dx);
        if (r + 32 < Mx_CACHE)
            __builtin_prefetch(cacheV + ((size_t)bh * Mx_CACHE + r + 32) * Dx, 0, 0);
        #pragma unroll
        for (int t = 0; t < 8; ++t) {
            v16bf bv;
            #pragma unroll
            for (int i = 0; i < 16; ++i) {
                int rr = cb + ((i >> 3) << 4) + koff + (i & 7);
                const float* vr = (rr >= P) ? (Vn + ((size_t)bh * Sx + (rr - P)) * Dx)
                                            : (cacheV + ((size_t)bh * Mx_CACHE + rr) * Dx);
                bv[i] = f2bf(vr[t * 16 + (lane & 15)]);
            }
            acc[t] = wmma_bf16(a, bv, acc[t]);
        }
    }
#endif

    int roff = (lane >> 4) << 3;
    #pragma unroll
    for (int t = 0; t < 8; ++t) {
        #pragma unroll
        for (int v = 0; v < 8; ++v) {
            int row = v + roff;
            if (row < Sx) atomicAdd(&o_sh[row * Dx + t * 16 + (lane & 15)], acc[t][v]);
        }
    }
    __syncthreads();

    int b = bh >> 4, h = bh & 15;
    for (int i = tid; i < Sx * Dx; i += 128) {
        int s = i >> 7, d = i & 127;
        out[((size_t)(b * Sx + s)) * Nx + h * Dx + d] = o_sh[i];
    }
}

// ---------------------------------------------------------------------------
// Workspace layout (bytes)
// ---------------------------------------------------------------------------
#define WS_XN   ((size_t)0)                       // 64*2048 u16      = 256 KB
#define WS_Q    ((size_t)262144)                  // 256*4*128 f32    = 512 KB
#define WS_KN   ((size_t)786432)                  // 512 KB
#define WS_VN   ((size_t)1310720)                 // 512 KB
#define WS_MX   ((size_t)1835008)                 // 1024 f32
#define WS_L    ((size_t)1839104)                 // 1024 f32
#define WS_S    ((size_t)1843200)                 // 256*4*4096 f32   = 16 MB

extern "C" void kernel_launch(void* const* d_in, const int* in_sizes, int n_in,
                              void* d_out, int out_size, void* d_ws, size_t ws_size,
                              hipStream_t stream) {
    const float* X  = (const float*)d_in[0];
    const float* Wq = (const float*)d_in[1];
    const float* Wk = (const float*)d_in[2];
    const float* Wv = (const float*)d_in[3];
    const float* cK = (const float*)d_in[4];
    const float* cV = (const float*)d_in[5];
    const int*   Pp = (const int*)d_in[6];

    char* ws = (char*)d_ws;
    unsigned short* Xn = (unsigned short*)(ws + WS_XN);
    float* Qws = (float*)(ws + WS_Q);
    float* Kn  = (float*)(ws + WS_KN);
    float* Vn  = (float*)(ws + WS_VN);
    float* Mxv = (float*)(ws + WS_MX);
    float* L   = (float*)(ws + WS_L);
    float* Sws = (float*)(ws + WS_S);
    float* out = (float*)d_out;

    rmsnorm_kernel<<<ROWS, 256, 0, stream>>>(X, Xn);
    init_mx_kernel<<<4, 256, 0, stream>>>(Mxv);
    qkv_kernel<<<dim3(Nx / 16, 3), 128, 0, stream>>>(Xn, Wq, Wk, Wv, Qws, Kn, Vn);
    scores_kernel<<<dim3(4, BH), 128, 0, stream>>>(Qws, cK, Kn, Pp, Sws, Mxv);
    sums_kernel<<<BH, 256, 0, stream>>>(Sws, Mxv, L);
    pv_kernel<<<BH, 128, 0, stream>>>(Sws, Mxv, L, cV, Vn, Pp, out);
}
